// HeuristicAdaptiveTTT_61761629716713
// MI455X (gfx1250) — matrix operand order repaired and checked by reading, not compiled, not run on hardware
//
#include <hip/hip_runtime.h>
#include <hip/hip_bf16.h>

// ---------------------------------------------------------------------------
// HeuristicAdaptiveTTT on MI455X (gfx1250, wave32, WMMA)
//
// Shapes (hardcoded from reference): B=2, S=2048, D=2048, V=32000.
// Math reduction:
//   grad = 2/(cnt*D) * G * (W - I),  G = (z*mask)^T (z*mask)   (symmetric)
//   W_{j+1} = W_j - 0.1*grad ;  out = sum_i (z*mask_i) @ W_final[i]
// GEMM core: v_wmma_f32_16x16x32_bf16, 128x128x32 block tiles, 8 waves,
// double-buffered LDS + register staging (1 barrier per K chunk),
// packed b64/b128 LDS stores for the f32->bf16 transpose path.
// ---------------------------------------------------------------------------

#define DD   2048
#define SS   2048
#define BBAT 2
#define VV   32000
#define NTOK (BBAT * SS)   // 4096
#define BK   32            // K chunk == one WMMA K step
#define PITCH 40           // bf16 elements per LDS row (32 + 8 pad, 16B multiple)
#define NCHUNK (DD / BK)   // 64 (same for S)

typedef __attribute__((ext_vector_type(16))) __bf16 v16bf;
typedef __attribute__((ext_vector_type(8)))  float  v8f;

union FragU { v16bf v; uint4 u[2]; };
union Pack4 { __bf16 h[4]; uint2 u; };
union Pack8 { __bf16 h[8]; uint4 u; };

// A fragment: LDS layout [m][k] (k contiguous). element j -> K = (j&7)+((j>>3)<<4)+8*half
__device__ __forceinline__ v16bf fragA(const __bf16* As, int m, int half) {
  FragU f;
  const __bf16* p = As + m * PITCH + (half << 3);
  f.u[0] = *(const uint4*)(p);        // K = 8h .. 8h+7        (j=0..7)
  f.u[1] = *(const uint4*)(p + 16);   // K = 16+8h .. 16+8h+7  (j=8..15)
  return f.v;
}
// B fragment: LDS layout [n][k] (k contiguous). element j -> K = (j&7)+((j>>3)<<4)+8*half
__device__ __forceinline__ v16bf fragB(const __bf16* Bs, int n, int half) {
  FragU f;
  const __bf16* p = Bs + n * PITCH + (half << 3);
  f.u[0] = *(const uint4*)(p);
  f.u[1] = *(const uint4*)(p + 16);
  return f.v;
}

// ---------------------------------------------------------------------------
// 1) Entropy per token (online softmax, single 524MB pass -> HBM bound)
// ---------------------------------------------------------------------------
__global__ __launch_bounds__(256) void entropy_kernel(
    const float* __restrict__ logits, float* __restrict__ diff)
{
  const int token = blockIdx.x;                 // 0..NTOK-1
  const float* lp = logits + (size_t)token * VV;
  const int tid = threadIdx.x;

  float m = lp[tid];                            // V (=32000) is a multiple of 256
  float s = 1.0f, w = 0.0f;                     // s = sum e^{l-m}, w = sum (l-m)e^{l-m}
  for (int i = tid + 256; i < VV; i += 256) {
    float l = lp[i];
    if (l > m) {
      float d  = m - l;                         // <= 0
      float ed = __expf(d);
      w = (w + d * s) * ed;                     // rebase w to new max
      s = s * ed + 1.0f;
      m = l;
    } else {
      float e = __expf(l - m);
      s += e;
      w += (l - m) * e;
    }
  }

  __shared__ float sm[256], ss[256], sw[256];
  sm[tid] = m; ss[tid] = s; sw[tid] = w;
  __syncthreads();
  for (int off = 128; off > 0; off >>= 1) {
    if (tid < off) {
      float m1 = sm[tid], s1 = ss[tid], w1 = sw[tid];
      float m2 = sm[tid + off], s2 = ss[tid + off], w2 = sw[tid + off];
      float M  = fmaxf(m1, m2);
      float e1 = __expf(m1 - M), e2 = __expf(m2 - M);
      sm[tid] = M;
      ss[tid] = s1 * e1 + s2 * e2;
      sw[tid] = (w1 + (m1 - M) * s1) * e1 + (w2 + (m2 - M) * s2) * e2;
    }
    __syncthreads();
  }
  if (tid == 0) {
    float ent = __logf(ss[0]) - sw[0] / ss[0];
    diff[token] = ent / __logf((float)VV);
  }
}

// ---------------------------------------------------------------------------
// 2) Joint quantile (bitonic sort of 4096 in LDS) + bucketize + counts
// ---------------------------------------------------------------------------
__global__ __launch_bounds__(1024) void bucketize_kernel(
    const float* __restrict__ diff, int* __restrict__ bucket,
    float* __restrict__ counts)
{
  const int N = NTOK;
  __shared__ float sbuf[NTOK];
  __shared__ int   scnt[6];
  __shared__ float th[2];
  const int tid = threadIdx.x;

  for (int i = tid; i < N; i += 1024) sbuf[i] = diff[i];
  if (tid < 6) scnt[tid] = 0;
  __syncthreads();

  for (int k = 2; k <= N; k <<= 1) {
    for (int j = k >> 1; j > 0; j >>= 1) {
      for (int i = tid; i < N; i += 1024) {
        int ixj = i ^ j;
        if (ixj > i) {
          float a = sbuf[i], b = sbuf[ixj];
          bool up = ((i & k) == 0);
          if ((a > b) == up) { sbuf[i] = b; sbuf[ixj] = a; }
        }
      }
      __syncthreads();
    }
  }
  if (tid < 2) {                                // jnp.quantile, linear interpolation
    float q   = (tid == 0) ? 0.3f : 0.7f;
    float pos = q * (float)(N - 1);
    int   lo  = (int)floorf(pos);
    float fr  = pos - (float)lo;
    th[tid] = sbuf[lo] * (1.0f - fr) + sbuf[lo + 1] * fr;
  }
  __syncthreads();
  float t0 = th[0], t1 = th[1];
  for (int i = tid; i < N; i += 1024) {
    float d = diff[i];
    int bkt = (d > t0 ? 1 : 0) + (d > t1 ? 1 : 0);
    bucket[i] = bkt;
    atomicAdd(&scnt[(i >> 11) * 3 + bkt], 1);   // i>>11 == batch (S=2048)
  }
  __syncthreads();
  if (tid < 6) counts[tid] = fmaxf((float)scnt[tid], 1.0f);
}

// ---------------------------------------------------------------------------
// 3) LayerNorm: z = (x-mu)*rsqrt(var+eps)*gamma + beta  (one block per token)
// ---------------------------------------------------------------------------
__global__ __launch_bounds__(256) void layernorm_kernel(
    const float* __restrict__ x, const float* __restrict__ gamma,
    const float* __restrict__ beta, float* __restrict__ z)
{
  const int token = blockIdx.x;
  const float* xr = x + (size_t)token * DD;
  float* zr = z + (size_t)token * DD;
  const int tid = threadIdx.x;

  float4 vals[2];
  float s = 0.0f, s2 = 0.0f;
#pragma unroll
  for (int q = 0; q < 2; ++q) {
    float4 v = ((const float4*)xr)[tid + q * 256];
    vals[q] = v;
    s  += v.x + v.y + v.z + v.w;
    s2 += v.x * v.x + v.y * v.y + v.z * v.z + v.w * v.w;
  }
  __shared__ float rs[256], rs2[256];
  rs[tid] = s; rs2[tid] = s2;
  __syncthreads();
  for (int off = 128; off > 0; off >>= 1) {
    if (tid < off) { rs[tid] += rs[tid + off]; rs2[tid] += rs2[tid + off]; }
    __syncthreads();
  }
  float mu  = rs[0] * (1.0f / DD);
  float var = rs2[0] * (1.0f / DD) - mu * mu;
  float inv = rsqrtf(var + 1e-5f);
#pragma unroll
  for (int q = 0; q < 2; ++q) {
    int i4 = tid + q * 256;
    float4 g = ((const float4*)gamma)[i4];
    float4 b = ((const float4*)beta)[i4];
    float4 v = vals[q], o;
    o.x = (v.x - mu) * inv * g.x + b.x;
    o.y = (v.y - mu) * inv * g.y + b.y;
    o.z = (v.z - mu) * inv * g.z + b.z;
    o.w = (v.w - mu) * inv * g.w + b.w;
    ((float4*)zr)[i4] = o;
  }
}

// ---------------------------------------------------------------------------
// 4) Gram: G[b,i] = (z*mask_i)^T z  (symmetric) -- upper-triangle blocks only,
//    mirrored on store. Double-buffered LDS, packed b64 transposed stores.
// ---------------------------------------------------------------------------
__global__ __launch_bounds__(256) void gram_kernel(
    const float* __restrict__ z, const int* __restrict__ bucket,
    float* __restrict__ G)
{
  if (blockIdx.x < blockIdx.y) return;          // symmetry: only nBase >= mBase
  const int pair = blockIdx.z;                  // b*3 + i
  const int b = pair / 3, buck = pair % 3;
  const int mBase = blockIdx.y * 128;
  const int nBase = blockIdx.x * 128;
  const float* zb = z + (size_t)b * SS * DD;
  const int* bk = bucket + b * SS;

  __shared__ __align__(16) __bf16 As[2][128 * PITCH];
  __shared__ __align__(16) __bf16 Bs[2][128 * PITCH];

  const int tid  = threadIdx.x;
  const int lane = tid & 31, wave = tid >> 5;
  const int half = lane >> 4, l16 = lane & 15;
  const int wm = (wave & 3) * 32;               // wave: 32 rows x 64 cols
  const int wn = (wave >> 2) * 64;

  // transpose loader: thread owns 4 tokens x 4 d
  const int trow0 = (tid >> 5) * 4;             // token offset in chunk (0..28)
  const int c0    = (tid & 31) * 4;             // d offset in tile (0..124)

  float sa[4][4], sb[4][4];
  v8f acc[2][4] = {};

  auto loadChunk = [&](int k0) {
#pragma unroll
    for (int t = 0; t < 4; ++t) {
      const int tok = k0 + trow0 + t;
      const float mk = (bk[tok] == buck) ? 1.0f : 0.0f;
      const float* r = zb + (size_t)tok * DD;
      float4 va = *(const float4*)(r + mBase + c0);
      float4 vb = *(const float4*)(r + nBase + c0);
      sa[t][0] = va.x * mk; sa[t][1] = va.y * mk;
      sa[t][2] = va.z * mk; sa[t][3] = va.w * mk;
      sb[t][0] = vb.x; sb[t][1] = vb.y; sb[t][2] = vb.z; sb[t][3] = vb.w;
    }
  };
  auto storeChunk = [&](int bi) {
#pragma unroll
    for (int c = 0; c < 4; ++c) {
      Pack4 pa, pb;
#pragma unroll
      for (int t = 0; t < 4; ++t) { pa.h[t] = (__bf16)sa[t][c]; pb.h[t] = (__bf16)sb[t][c]; }
      *(uint2*)&As[bi][(c0 + c) * PITCH + trow0] = pa.u;
      *(uint2*)&Bs[bi][(c0 + c) * PITCH + trow0] = pb.u;
    }
  };

  loadChunk(0);
  storeChunk(0);
  __syncthreads();
  for (int c = 0; c < NCHUNK; ++c) {
    const int cur = c & 1;
    if (c + 1 < NCHUNK) loadChunk((c + 1) * BK);   // global loads overlap WMMAs
    v16bf af[2], bf[4];
#pragma unroll
    for (int i = 0; i < 2; ++i) af[i] = fragA(As[cur], wm + i * 16 + l16, half);
#pragma unroll
    for (int j = 0; j < 4; ++j) bf[j] = fragB(Bs[cur], wn + j * 16 + l16, half);
#pragma unroll
    for (int i = 0; i < 2; ++i)
#pragma unroll
      for (int j = 0; j < 4; ++j)
        acc[i][j] = __builtin_amdgcn_wmma_f32_16x16x32_bf16(
            false, af[i], false, bf[j], (short)0, acc[i][j], false, false);
    if (c + 1 < NCHUNK) storeChunk(cur ^ 1);
    __syncthreads();
  }

  float* Gp = G + (size_t)pair * DD * DD;
  const bool mirror = (blockIdx.x != blockIdx.y);
#pragma unroll
  for (int i = 0; i < 2; ++i)
#pragma unroll
    for (int j = 0; j < 4; ++j)
#pragma unroll
      for (int v = 0; v < 8; ++v) {
        int m = mBase + wm + i * 16 + v + 8 * half;
        int n = nBase + wn + j * 16 + l16;
        float val = acc[i][j][v];
        Gp[(size_t)m * DD + n] = val;
        if (mirror) Gp[(size_t)n * DD + m] = val;
      }
}

// ---------------------------------------------------------------------------
// 5) TTT update: Wout = Win - alpha * ( G @ (Win - I) ),  alpha = 0.2/(cnt*D)
//    G symmetric -> A direct layout; B transposed with packed b64 stores.
// ---------------------------------------------------------------------------
struct PairMap { int p[6]; };

__global__ __launch_bounds__(256) void ttt_update_kernel(
    const float* __restrict__ G, const float* __restrict__ Win,
    long long wstride, float* __restrict__ Wout,
    const float* __restrict__ counts, PairMap map)
{
  const int slot = map.p[blockIdx.z];
  const float* Gp = G + (size_t)slot * DD * DD;
  const float* Wi = Win + (size_t)slot * (size_t)wstride;  // wstride 0 => shared W0
  float* Wo = Wout + (size_t)slot * DD * DD;
  const float alpha = 0.2f / (counts[slot] * (float)DD);

  const int mBase = blockIdx.y * 128;
  const int nBase = blockIdx.x * 128;

  __shared__ __align__(16) __bf16 As[2][128 * PITCH];
  __shared__ __align__(16) __bf16 Bs[2][128 * PITCH];

  const int tid  = threadIdx.x;
  const int lane = tid & 31, wave = tid >> 5;
  const int half = lane >> 4, l16 = lane & 15;
  const int wm = (wave & 3) * 32;
  const int wn = (wave >> 2) * 64;

  // direct A loader: thread owns 1 row x 16 k
  const int arow  = tid >> 1;
  const int apart = (tid & 1) << 4;
  // transposed B loader: thread owns 4 k x 4 n
  const int krow0 = (tid >> 5) * 4;
  const int c0    = (tid & 31) * 4;

  float4 sA[4];
  float  sB[4][4];
  v8f acc[2][4] = {};

  auto loadChunk = [&](int k0) {
    const float* ap = Gp + (size_t)(mBase + arow) * DD + k0 + apart;
#pragma unroll
    for (int q = 0; q < 4; ++q) sA[q] = ((const float4*)ap)[q];
#pragma unroll
    for (int t = 0; t < 4; ++t) {
      const int gk = k0 + krow0 + t;
      float4 v = *(const float4*)(Wi + (size_t)gk * DD + nBase + c0);
      sB[t][0] = v.x - ((gk == nBase + c0 + 0) ? 1.0f : 0.0f);
      sB[t][1] = v.y - ((gk == nBase + c0 + 1) ? 1.0f : 0.0f);
      sB[t][2] = v.z - ((gk == nBase + c0 + 2) ? 1.0f : 0.0f);
      sB[t][3] = v.w - ((gk == nBase + c0 + 3) ? 1.0f : 0.0f);
    }
  };
  auto storeChunk = [&](int bi) {
    Pack8 p0, p1;
#pragma unroll
    for (int e = 0; e < 4; ++e) {
      p0.h[e]     = (__bf16)(((const float*)&sA[0])[e]);
      p0.h[e + 4] = (__bf16)(((const float*)&sA[1])[e]);
      p1.h[e]     = (__bf16)(((const float*)&sA[2])[e]);
      p1.h[e + 4] = (__bf16)(((const float*)&sA[3])[e]);
    }
    *(uint4*)&As[bi][arow * PITCH + apart]     = p0.u;
    *(uint4*)&As[bi][arow * PITCH + apart + 8] = p1.u;
#pragma unroll
    for (int c = 0; c < 4; ++c) {
      Pack4 pb;
#pragma unroll
      for (int t = 0; t < 4; ++t) pb.h[t] = (__bf16)sB[t][c];
      *(uint2*)&Bs[bi][(c0 + c) * PITCH + krow0] = pb.u;
    }
  };

  loadChunk(0);
  storeChunk(0);
  __syncthreads();
  for (int c = 0; c < NCHUNK; ++c) {
    const int cur = c & 1;
    if (c + 1 < NCHUNK) loadChunk((c + 1) * BK);
    v16bf af[2], bf[4];
#pragma unroll
    for (int i = 0; i < 2; ++i) af[i] = fragA(As[cur], wm + i * 16 + l16, half);
#pragma unroll
    for (int j = 0; j < 4; ++j) bf[j] = fragB(Bs[cur], wn + j * 16 + l16, half);
#pragma unroll
    for (int i = 0; i < 2; ++i)
#pragma unroll
      for (int j = 0; j < 4; ++j)
        acc[i][j] = __builtin_amdgcn_wmma_f32_16x16x32_bf16(
            false, af[i], false, bf[j], (short)0, acc[i][j], false, false);
    if (c + 1 < NCHUNK) storeChunk(cur ^ 1);
    __syncthreads();
  }

#pragma unroll
  for (int i = 0; i < 2; ++i)
#pragma unroll
    for (int j = 0; j < 4; ++j)
#pragma unroll
      for (int v = 0; v < 8; ++v) {
        int m = mBase + wm + i * 16 + v + 8 * half;
        int n = nBase + wn + j * 16 + l16;
        size_t idx = (size_t)m * DD + n;
        Wo[idx] = Wi[idx] - alpha * acc[i][j][v];
      }
}

// ---------------------------------------------------------------------------
// 6) Output: out = sum_i (z*mask_i) @ W_final[b,i]   (mask partitions tokens)
// ---------------------------------------------------------------------------
__global__ __launch_bounds__(256) void output_kernel(
    const float* __restrict__ z, const int* __restrict__ bucket,
    const float* __restrict__ WbufA, const float* __restrict__ WbufB,
    float* __restrict__ out)
{
  const int b = blockIdx.z;
  const int tBase = blockIdx.y * 128;           // token block within batch
  const int nBase = blockIdx.x * 128;
  const float* zb = z + (size_t)b * SS * DD;
  const int* bk = bucket + b * SS;

  __shared__ __align__(16) __bf16 As[2][128 * PITCH];
  __shared__ __align__(16) __bf16 Bs[2][128 * PITCH];

  const int tid  = threadIdx.x;
  const int lane = tid & 31, wave = tid >> 5;
  const int half = lane >> 4, l16 = lane & 15;
  const int wm = (wave & 3) * 32;
  const int wn = (wave >> 2) * 64;

  const int arow  = tid >> 1;
  const int apart = (tid & 1) << 4;
  const int krow0 = (tid >> 5) * 4;
  const int c0    = (tid & 31) * 4;

  float4 sA[4];
  float  sB[4][4];
  v8f acc[2][4] = {};

  const int tok = tBase + arow;

  for (int ibk = 0; ibk < 3; ++ibk) {
    // finals: bucket0 in bufA (1 step), buckets 1,2 in bufB (2 and 4 steps)
    const float* W = ((ibk == 0) ? WbufA : WbufB) + (size_t)(b * 3 + ibk) * DD * DD;
    const float msk = (bk[tok] == ibk) ? 1.0f : 0.0f;

    auto loadChunk = [&](int k0) {
      const float* ap = zb + (size_t)tok * DD + k0 + apart;
#pragma unroll
      for (int q = 0; q < 4; ++q) {
        float4 v = ((const float4*)ap)[q];
        v.x *= msk; v.y *= msk; v.z *= msk; v.w *= msk;
        sA[q] = v;
      }
#pragma unroll
      for (int t = 0; t < 4; ++t) {
        float4 v = *(const float4*)(W + (size_t)(k0 + krow0 + t) * DD + nBase + c0);
        sB[t][0] = v.x; sB[t][1] = v.y; sB[t][2] = v.z; sB[t][3] = v.w;
      }
    };
    auto storeChunk = [&](int bi) {
      Pack8 p0, p1;
#pragma unroll
      for (int e = 0; e < 4; ++e) {
        p0.h[e]     = (__bf16)(((const float*)&sA[0])[e]);
        p0.h[e + 4] = (__bf16)(((const float*)&sA[1])[e]);
        p1.h[e]     = (__bf16)(((const float*)&sA[2])[e]);
        p1.h[e + 4] = (__bf16)(((const float*)&sA[3])[e]);
      }
      *(uint4*)&As[bi][arow * PITCH + apart]     = p0.u;
      *(uint4*)&As[bi][arow * PITCH + apart + 8] = p1.u;
#pragma unroll
      for (int c = 0; c < 4; ++c) {
        Pack4 pb;
#pragma unroll
        for (int t = 0; t < 4; ++t) pb.h[t] = (__bf16)sB[t][c];
        *(uint2*)&Bs[bi][(c0 + c) * PITCH + krow0] = pb.u;
      }
    };

    loadChunk(0);
    storeChunk(0);
    __syncthreads();
    for (int c = 0; c < NCHUNK; ++c) {
      const int cur = c & 1;
      if (c + 1 < NCHUNK) loadChunk((c + 1) * BK);
      v16bf af[2], bf[4];
#pragma unroll
      for (int i = 0; i < 2; ++i) af[i] = fragA(As[cur], wm + i * 16 + l16, half);
#pragma unroll
      for (int j = 0; j < 4; ++j) bf[j] = fragB(Bs[cur], wn + j * 16 + l16, half);
#pragma unroll
      for (int i = 0; i < 2; ++i)
#pragma unroll
        for (int j = 0; j < 4; ++j)
          acc[i][j] = __builtin_amdgcn_wmma_f32_16x16x32_bf16(
              false, af[i], false, bf[j], (short)0, acc[i][j], false, false);
      if (c + 1 < NCHUNK) storeChunk(cur ^ 1);
      __syncthreads();
    }
  }

#pragma unroll
  for (int i = 0; i < 2; ++i)
#pragma unroll
    for (int j = 0; j < 4; ++j)
#pragma unroll
      for (int v = 0; v < 8; ++v) {
        int m = tBase + wm + i * 16 + v + 8 * half;   // token index in batch
        int n = nBase + wn + j * 16 + l16;
        out[((size_t)b * SS + m) * DD + n] = acc[i][j][v];
      }
}

// ---------------------------------------------------------------------------
extern "C" void kernel_launch(void* const* d_in, const int* in_sizes, int n_in,
                              void* d_out, int out_size, void* d_ws, size_t ws_size,
                              hipStream_t stream)
{
  const float* x      = (const float*)d_in[0];
  const float* logits = (const float*)d_in[1];
  const float* W0     = (const float*)d_in[2];
  const float* gamma  = (const float*)d_in[3];
  const float* beta   = (const float*)d_in[4];
  float* out = (float*)d_out;

  // Workspace carve-up (~336 MB): diff | bucket | counts | z | G[6] | bufA[6] | bufB[6]
  char* ws = (char*)d_ws;
  size_t off = 0;
  auto carve = [&](size_t bytes) -> void* {
    void* p = ws + off;
    off += (bytes + 255) & ~(size_t)255;
    return p;
  };
  const size_t MAT = (size_t)DD * DD;           // elements per DxD matrix
  float* diff   = (float*)carve((size_t)NTOK * 4);
  int*   bucket = (int*)  carve((size_t)NTOK * 4);
  float* counts = (float*)carve(6 * 4);
  float* z      = (float*)carve((size_t)BBAT * SS * DD * 4);
  float* G      = (float*)carve(6 * MAT * 4);
  float* bufA   = (float*)carve(6 * MAT * 4);
  float* bufB   = (float*)carve(6 * MAT * 4);

  entropy_kernel  <<<NTOK, 256, 0, stream>>>(logits, diff);
  bucketize_kernel<<<1, 1024, 0, stream>>>(diff, bucket, counts);
  layernorm_kernel<<<NTOK, 256, 0, stream>>>(x, gamma, beta, z);

  dim3 gtile(DD / 128, DD / 128);               // 16 x 16
  gram_kernel<<<dim3(gtile.x, gtile.y, 6), 256, 0, stream>>>(z, bucket, G);

  // Fixed-point iterations, ping-pong; buckets run k = (1,2,4) steps.
  PairMap all = {{0, 1, 2, 3, 4, 5}};           // step 0: every pair
  PairMap mid = {{1, 2, 4, 5, 0, 0}};           // step 1: buckets 1,2
  PairMap top = {{2, 5, 0, 0, 0, 0}};           // steps 2,3: bucket 2
  ttt_update_kernel<<<dim3(gtile.x, gtile.y, 6), 256, 0, stream>>>(
      G, W0,   0LL,            bufA, counts, all);   // W0 -> bufA
  ttt_update_kernel<<<dim3(gtile.x, gtile.y, 4), 256, 0, stream>>>(
      G, bufA, (long long)MAT, bufB, counts, mid);   // bufA -> bufB
  ttt_update_kernel<<<dim3(gtile.x, gtile.y, 2), 256, 0, stream>>>(
      G, bufB, (long long)MAT, bufA, counts, top);   // bufB -> bufA
  ttt_update_kernel<<<dim3(gtile.x, gtile.y, 2), 256, 0, stream>>>(
      G, bufA, (long long)MAT, bufB, counts, top);   // bufA -> bufB
  // Finals: bucket0 slots in bufA; buckets 1 and 2 slots in bufB.

  output_kernel<<<dim3(DD / 128, SS / 128, BBAT), 256, 0, stream>>>(
      z, bucket, bufA, bufB, out);
}
